// LSTMCell_57647051047076
// MI455X (gfx1250) — matrix operand order repaired and checked by reading, not compiled, not run on hardware
//
#include <hip/hip_runtime.h>
#include <hip/hip_bf16.h>

typedef _Float16 v4h  __attribute__((ext_vector_type(4)));
typedef _Float16 v8h  __attribute__((ext_vector_type(8)));
typedef _Float16 v16h __attribute__((ext_vector_type(16)));
typedef float    v8f  __attribute__((ext_vector_type(8)));

#define BSZ  16384
#define HN   1024
#define KTOT 2048                       // concatenated K: [x | hidden]
#define NK   64                         // 64 k-steps of 32

#define BM 128
#define BN 64
#define LDS_STRIDE 40                   // halves per row (80 B, bank friendly)
#define A_TILE_H (BM * LDS_STRIDE)      // 5120 halves
#define BUF_H    (A_TILE_H + 4 * BN * LDS_STRIDE)  // 15360 halves / buffer
#define BUF_B    (BUF_H * 2)            // 30720 bytes / buffer
#define NBUF 4

// -------- f32 -> f16 packing: src rows of 1024 f32 -> dst rows of 2048 f16 --------
__global__ void cvt_pack(const float* __restrict__ src, _Float16* __restrict__ dst,
                         int n, int col4off) {
  int stride = gridDim.x * blockDim.x;
  int n4 = n >> 2;
  for (int i = blockIdx.x * blockDim.x + threadIdx.x; i < n4; i += stride) {
    int row = i >> 8;                   // 256 float4 per 1024-wide src row
    int col = i & 255;
    float4 v = ((const float4*)src)[i];
    v4h h;
    h[0] = (_Float16)v.x; h[1] = (_Float16)v.y;
    h[2] = (_Float16)v.z; h[3] = (_Float16)v.w;
    ((v4h*)dst)[(size_t)row * 512 + col4off + col] = h;   // 512 v4h per dst row
  }
}

__device__ __forceinline__ float sigm(float x) {
  return 1.0f / (1.0f + __expf(-x));
}

// -------- fused 8-GEMM + LSTM epilogue --------
// Acat: [B][2048] f16 (x|hidden), Wcat: [gate f,i,c,o][H][2048] f16 (Wgx|Wgh)
__launch_bounds__(256)
__global__ void lstm_fused(const _Float16* __restrict__ Acat,
                           const _Float16* __restrict__ Wcat,
                           const float* __restrict__ bf, const float* __restrict__ bi,
                           const float* __restrict__ bc, const float* __restrict__ bo,
                           const float* __restrict__ cell,
                           float* __restrict__ out_h, float* __restrict__ out_c) {
  extern __shared__ _Float16 lds[];     // NBUF * BUF_B bytes

  const int t    = threadIdx.x;
  const int bN   = blockIdx.x * BN;
  const int bM   = blockIdx.y * BM;
  const int lane = t & 31;
  const int wave = t >> 5;
  const int wm   = wave >> 2;           // 0..1
  const int wn   = wave & 3;            // 0..3
  const int l16  = lane >> 4;
  const int lr   = lane & 15;

  // staging map: thread -> (row = t/4, 16B chunk = t%4)
  const int arow  = t >> 2;             // 0..63
  const int achkB = (t & 3) * 16;       // byte offset within 64B row slice

  // ---- uniform SGPR bases (advance 64B per k-step) ----
  const _Float16* baseA = Acat + (size_t)bM * KTOT;
  const _Float16* baseW = Wcat + (size_t)bN * KTOT;

  // ---- loop-invariant per-lane byte offsets (GVS vaddr, i32) ----
  const int vA0 = arow * (KTOT * 2) + achkB;
  const int vA1 = vA0 + 64 * (KTOT * 2);
  const int vW0 = arow * (KTOT * 2) + achkB;
  const int vW1 = vW0 + HN * (KTOT * 2);
  const int vW2 = vW1 + HN * (KTOT * 2);
  const int vW3 = vW2 + HN * (KTOT * 2);

  // ---- LDS destination byte addresses (within buffer 0) ----
  const unsigned ldsBase = (unsigned)(size_t)(&lds[0]);
  const unsigned lA0 = ldsBase + (unsigned)(arow * 80 + achkB);
  const unsigned lA1 = lA0 + 64 * 80;
  const unsigned lB0 = ldsBase + (unsigned)(A_TILE_H * 2 + arow * 80 + achkB);
  const unsigned lB1 = lB0 + 64 * 80;
  const unsigned lB2 = lB1 + 64 * 80;
  const unsigned lB3 = lB2 + 64 * 80;

  v8f acc[4][4];                        // [gate][m-subtile]
  #pragma unroll
  for (int g = 0; g < 4; ++g)
    #pragma unroll
    for (int m = 0; m < 4; ++m) acc[g][m] = v8f{};

  unsigned ibuf = 0;                    // issue-side buffer byte offset
  auto issue6 = [&]() {
    unsigned long long bA = (unsigned long long)(size_t)baseA;
    unsigned long long bW = (unsigned long long)(size_t)baseW;
    asm volatile("global_load_async_to_lds_b128 %0, %1, %2"
                 :: "v"(lA0 + ibuf), "v"(vA0), "s"(bA) : "memory");
    asm volatile("global_load_async_to_lds_b128 %0, %1, %2"
                 :: "v"(lA1 + ibuf), "v"(vA1), "s"(bA) : "memory");
    asm volatile("global_load_async_to_lds_b128 %0, %1, %2"
                 :: "v"(lB0 + ibuf), "v"(vW0), "s"(bW) : "memory");
    asm volatile("global_load_async_to_lds_b128 %0, %1, %2"
                 :: "v"(lB1 + ibuf), "v"(vW1), "s"(bW) : "memory");
    asm volatile("global_load_async_to_lds_b128 %0, %1, %2"
                 :: "v"(lB2 + ibuf), "v"(vW2), "s"(bW) : "memory");
    asm volatile("global_load_async_to_lds_b128 %0, %1, %2"
                 :: "v"(lB3 + ibuf), "v"(vW3), "s"(bW) : "memory");
    baseA += 32;                        // +64 bytes per k-step
    baseW += 32;
    ibuf = (ibuf == (NBUF - 1) * BUF_B) ? 0u : ibuf + BUF_B;
  };

  unsigned cbufH = 0;                   // compute-side buffer offset (halves)
  auto compute = [&]() {
    const _Float16* Ab = lds + cbufH;
    const _Float16* Bb = Ab + A_TILE_H;
    v16h af[4];
    #pragma unroll
    for (int m = 0; m < 4; ++m) {
      const int row = wm * 64 + m * 16 + lr;
      v8h c0 = *(const v8h*)(Ab + row * LDS_STRIDE + l16 * 8);
      v8h c1 = *(const v8h*)(Ab + row * LDS_STRIDE + 16 + l16 * 8);
      af[m] = __builtin_shufflevector(c0, c1, 0,1,2,3,4,5,6,7,8,9,10,11,12,13,14,15);
    }
    #pragma unroll
    for (int g = 0; g < 4; ++g) {
      const int brow = g * 64 + wn * 16 + lr;
      v8h d0 = *(const v8h*)(Bb + brow * LDS_STRIDE + l16 * 8);
      v8h d1 = *(const v8h*)(Bb + brow * LDS_STRIDE + 16 + l16 * 8);
      v16h bfr = __builtin_shufflevector(d0, d1, 0,1,2,3,4,5,6,7,8,9,10,11,12,13,14,15);
      #pragma unroll
      for (int m = 0; m < 4; ++m)
        acc[g][m] = __builtin_amdgcn_wmma_f32_16x16x32_f16(
            false, af[m], false, bfr, (short)0, acc[g][m], false, false);
    }
    cbufH = (cbufH == (NBUF - 1) * BUF_H) ? 0u : cbufH + BUF_H;
  };

  // ---- pipeline: prefetch distance 2, 4 LDS buffers, ONE barrier/iter ----
  // WAR safety: issue(kk+2) writes buf (kk+2)%4 = buf (kk-2)%4, whose readers
  // (compute(kk-2)) are strictly before the barrier of iteration kk-1 in every
  // wave's program order, and that barrier precedes this issue.
  issue6();                             // tile 0
  issue6();                             // tile 1
  for (int kk = 0; kk < NK - 2; ++kk) {
    issue6();                           // tile kk+2
    asm volatile("s_wait_asynccnt 0xc" ::: "memory");   // tile kk complete
    __syncthreads();
    compute();                          // reads buffer kk%4
  }
  // tail: kk = NK-2
  asm volatile("s_wait_asynccnt 0x6" ::: "memory");
  __syncthreads();
  compute();
  // tail: kk = NK-1
  asm volatile("s_wait_asynccnt 0x0" ::: "memory");
  __syncthreads();
  compute();

  // ---- epilogue: bias + activations + cell update ----
  const int gn = bN + wn * 16 + lr;
  const float bfv = bf[gn], biv = bi[gn], bcv = bc[gn], bov = bo[gn];
  #pragma unroll
  for (int m = 0; m < 4; ++m) {
    const int gmBase = bM + wm * 64 + m * 16 + l16 * 8;
    #pragma unroll
    for (int r = 0; r < 8; ++r) {
      const int gm = gmBase + r;
      const size_t idx = (size_t)gm * HN + gn;
      float fg = sigm(acc[0][m][r] + bfv);
      float ig = sigm(acc[1][m][r] + biv);
      float cg = tanhf(acc[2][m][r] + bcv);
      float og = sigm(acc[3][m][r] + bov);
      float cn = fg * cell[idx] + ig * cg;
      out_h[idx] = og * tanhf(cn);
      out_c[idx] = cn;
    }
  }
}

extern "C" void kernel_launch(void* const* d_in, const int* in_sizes, int n_in,
                              void* d_out, int out_size, void* d_ws, size_t ws_size,
                              hipStream_t stream) {
  const size_t BI  = (size_t)BSZ * 1024;     // x / hidden element counts
  const size_t WSZ = (size_t)HN * 1024;      // per weight matrix

  _Float16* Acat = (_Float16*)d_ws;                      // [B][2048]
  _Float16* Wcat = Acat + (size_t)BSZ * KTOT;            // [4][H][2048]

  // pack x | hidden
  cvt_pack<<<2048, 256, 0, stream>>>((const float*)d_in[0], Acat, (int)BI, 0);
  cvt_pack<<<2048, 256, 0, stream>>>((const float*)d_in[1], Acat, (int)BI, 256);
  // pack weights: gates f,i,c,o; x-part cols 0..1023, h-part cols 1024..2047
  const int wx[4] = {3, 6, 9, 12};
  const int wh[4] = {5, 8, 11, 14};
  for (int g = 0; g < 4; ++g) {
    _Float16* dst = Wcat + (size_t)g * HN * KTOT;
    cvt_pack<<<512, 256, 0, stream>>>((const float*)d_in[wx[g]], dst, (int)WSZ, 0);
    cvt_pack<<<512, 256, 0, stream>>>((const float*)d_in[wh[g]], dst, (int)WSZ, 256);
  }

  float* out_h = (float*)d_out;
  float* out_c = out_h + (size_t)BSZ * HN;

  dim3 grid(HN / BN, BSZ / BM);   // 16 x 128 blocks
  lstm_fused<<<grid, 256, NBUF * BUF_B, stream>>>(
      Acat, Wcat,
      (const float*)d_in[4], (const float*)d_in[7],
      (const float*)d_in[10], (const float*)d_in[13],
      (const float*)d_in[2], out_h, out_c);
}
